// QuantizedLinear_42580305772730
// MI455X (gfx1250) — 8-bit matrix operand order repaired and checked by reading, compiled, not run on hardware
//
#include <hip/hip_runtime.h>

typedef __attribute__((ext_vector_type(8))) int v8i;
typedef __attribute__((ext_vector_type(4))) int v4i;

#define M_TOK 8192
#define K_FEAT 4096
#define N_OUT 4096

#define BM 128
#define BN 128
#define BK 64
#define LDS_STRIDE 80
#define BUFSZ (BM * LDS_STRIDE)

#ifndef __has_builtin
#define __has_builtin(x) 0
#endif

#if __has_builtin(__builtin_amdgcn_global_load_async_to_lds_b128)
#define HAVE_ASYNC_LDS 1
#else
#define HAVE_ASYNC_LDS 0
#endif
#if __has_builtin(__builtin_amdgcn_s_wait_asynccnt)
#define HAVE_WAIT_ASYNC_BUILTIN 1
#else
#define HAVE_WAIT_ASYNC_BUILTIN 0
#endif

#define AS1 __attribute__((address_space(1)))
#define AS3 __attribute__((address_space(3)))

__global__ __launch_bounds__(256)
void pack_i32_to_i8(const int* __restrict__ src, unsigned int* __restrict__ dst, int n4) {
    int i = blockIdx.x * blockDim.x + threadIdx.x;
    if (i < n4) {
        const int4 v = ((const int4*)src)[i];
        unsigned int p = (v.x & 0xFF) | ((v.y & 0xFF) << 8) |
                         ((v.z & 0xFF) << 16) | ((unsigned)(v.w & 0xFF) << 24);
        dst[i] = p;
    }
}

__device__ __forceinline__ void wait_async_done() {
#if HAVE_ASYNC_LDS
#if HAVE_WAIT_ASYNC_BUILTIN
    __builtin_amdgcn_s_wait_asynccnt(0);
#else
    asm volatile("s_wait_asynccnt 0" ::: "memory");
#endif
#endif
}

__device__ __forceinline__ void copy16_to_lds(const unsigned char* g, unsigned char* l) {
#if HAVE_ASYNC_LDS
    __builtin_amdgcn_global_load_async_to_lds_b128(
        (AS1 v4i*)g, (AS3 v4i*)l,  0,  0);
#else
    *(int4*)l = *(const int4*)g;
#endif
}

__device__ __forceinline__ void stage_tile(const unsigned char* __restrict__ xq,
                                           const unsigned char* __restrict__ wq,
                                           unsigned char* Abuf, unsigned char* Bbuf,
                                           int tid, int blockM, int blockN, int k0) {
#pragma unroll
    for (int i = 0; i < 2; ++i) {
        int c   = tid + i * 256;
        int row = c >> 2;
        int col = (c & 3) << 4;
        copy16_to_lds(xq + (size_t)(blockM + row) * K_FEAT + k0 + col,
                      Abuf + row * LDS_STRIDE + col);
        copy16_to_lds(wq + (size_t)(blockN + row) * K_FEAT + k0 + col,
                      Bbuf + row * LDS_STRIDE + col);
    }
}

__global__ __launch_bounds__(256)
void qlinear_wmma(const unsigned char* __restrict__ xq,
                  const unsigned char* __restrict__ wq,
                  const int*   __restrict__ bias,
                  const float* __restrict__ wscale,
                  float* __restrict__ out)
{
    __shared__ unsigned char As[2 * BUFSZ];
    __shared__ unsigned char Bs[2 * BUFSZ];

    const int tid   = threadIdx.x;
    const int lane  = tid & 31;
    const int wid   = tid >> 5;
    const int waveM = wid & 1;
    const int waveN = wid >> 1;
    const int half  = lane >> 4;
    const int l16   = lane & 15;

    const int blockM = blockIdx.y * BM;
    const int blockN = blockIdx.x * BN;

    int ncol[2];
    ncol[0] = blockN + waveN * 32 + l16;
    ncol[1] = ncol[0] + 16;

    int   bval[2];
    float rq[2];
#pragma unroll
    for (int t = 0; t < 2; ++t) {
        bval[t] = bias[ncol[t]];
        rq[t]   = (0.02f / 0.05f) * wscale[ncol[t]];
    }

    v8i acc[4][2];
#pragma unroll
    for (int mt = 0; mt < 4; ++mt)
#pragma unroll
        for (int nt = 0; nt < 2; ++nt) {
            v8i c;
#pragma unroll
            for (int e = 0; e < 8; ++e) c[e] = bval[nt];
            acc[mt][nt] = c;
        }

    const int aFragBase = (waveM * 64 + l16) * LDS_STRIDE + half * 8;
    const int bFragBase = (waveN * 32 + l16) * LDS_STRIDE + half * 8;

    constexpr int NT = K_FEAT / BK;
    stage_tile(xq, wq, As, Bs, tid, blockM, blockN, 0);

#pragma unroll 1
    for (int kt = 0; kt < NT; ++kt) {
        const int co = (kt & 1) * BUFSZ;
        const int no = BUFSZ - co;
        const int knext = (kt + 1 < NT) ? (kt + 1) * BK : kt * BK;

        wait_async_done();
        __syncthreads();
        stage_tile(xq, wq, As + no, Bs + no, tid, blockM, blockN, knext);

        v8i bfrag[2];
#pragma unroll
        for (int nt = 0; nt < 2; ++nt) {
            const unsigned char* bp = &Bs[co + bFragBase + nt * 16 * LDS_STRIDE];
            int2 b0 = *(const int2*)(bp);
            int2 b1 = *(const int2*)(bp + 16);
            int2 b2 = *(const int2*)(bp + 32);
            int2 b3 = *(const int2*)(bp + 48);
            v8i b;
            b[0] = b0.x; b[1] = b0.y; b[2] = b1.x; b[3] = b1.y;
            b[4] = b2.x; b[5] = b2.y; b[6] = b3.x; b[7] = b3.y;
            bfrag[nt] = b;
        }

#pragma unroll
        for (int mt = 0; mt < 4; ++mt) {
            const unsigned char* ap = &As[co + aFragBase + mt * 16 * LDS_STRIDE];
            int2 a0 = *(const int2*)(ap);
            int2 a1 = *(const int2*)(ap + 16);
            int2 a2 = *(const int2*)(ap + 32);
            int2 a3 = *(const int2*)(ap + 48);
            v8i a;
            a[0] = a0.x; a[1] = a0.y; a[2] = a1.x; a[3] = a1.y;
            a[4] = a2.x; a[5] = a2.y; a[6] = a3.x; a[7] = a3.y;
#pragma unroll
            for (int nt = 0; nt < 2; ++nt) {
                acc[mt][nt] = __builtin_amdgcn_wmma_i32_16x16x64_iu8(
                    true, a, true, bfrag[nt], acc[mt][nt], false, false);
            }
        }
    }

#pragma unroll
    for (int mt = 0; mt < 4; ++mt) {
#pragma unroll
        for (int nt = 0; nt < 2; ++nt) {
#pragma unroll
            for (int r = 0; r < 8; ++r) {
                int m = blockM + waveM * 64 + mt * 16 + r + half * 8;
                float f = (float)acc[mt][nt][r] * rq[nt] + 5.0f;
                f = rintf(f);
                f = fminf(fmaxf(f, -128.0f), 127.0f);
                out[(size_t)m * N_OUT + ncol[nt]] = f;
            }
        }
    }
}

extern "C" void kernel_launch(void* const* d_in, const int* in_sizes, int n_in,
                              void* d_out, int out_size, void* d_ws, size_t ws_size,
                              hipStream_t stream) {
    const int*   x_i32  = (const int*)d_in[0];
    const int*   w_i32  = (const int*)d_in[1];
    const int*   bias   = (const int*)d_in[2];
    const float* wscale = (const float*)d_in[3];
    float*       out    = (float*)d_out;

    unsigned char* xq = (unsigned char*)d_ws;
    unsigned char* wq = xq + (size_t)M_TOK * K_FEAT;

    const int nx4 = M_TOK * K_FEAT / 4;
    const int nw4 = N_OUT * K_FEAT / 4;
    pack_i32_to_i8<<<nx4 / 256, 256, 0, stream>>>(x_i32, (unsigned int*)xq, nx4);
    pack_i32_to_i8<<<nw4 / 256, 256, 0, stream>>>(w_i32, (unsigned int*)wq, nw4);

    dim3 grid(N_OUT / BN, M_TOK / BM);
    qlinear_wmma<<<grid, 256, 0, stream>>>(xq, wq, bias, wscale, out);
}
